// AttentionBlock_61014305407580
// MI455X (gfx1250) — matrix operand order repaired and checked by reading, not compiled, not run on hardware
//
#include <hip/hip_runtime.h>

typedef _Float16 f16;
typedef __attribute__((ext_vector_type(16))) _Float16 v16h;
typedef __attribute__((ext_vector_type(8)))  _Float16 v8h;
typedef __attribute__((ext_vector_type(8)))  float    v8f;
typedef __attribute__((ext_vector_type(2)))  float    f2;
typedef __attribute__((ext_vector_type(4)))  float    f4;

#define B_    4
#define S_    2048
#define H_    16
#define D_    64
#define HD    (H_ * D_)
#define QTILE 64            // queries per workgroup (16 per wave)
#define KBLK  64            // keys staged per iteration

#if __has_builtin(__builtin_amdgcn_exp2f)
#define EXP2(x) __builtin_amdgcn_exp2f(x)
#else
#define EXP2(x) exp2f(x)
#endif

// Hardware A-fragment half h maps to hw-k = lg*8 + (h&7) + 16*(h>>3).
// We permute the contraction axis with pi(16*lg + h) = 8*lg + (h&7) + 16*(h>>3)
// (applied consistently to the A side we stage and the B side):
//  * A-side composed pattern becomes:  k_actual = 16*lg + h   (16 contiguous!)
//  * B-side (P^T) becomes PURELY LOCAL per lane: no cross-lane exchange.

union Frag16 { v16h v; v8h h[2]; };

__device__ __forceinline__ v8h cvt8(f4 a, f4 b) {
    v8h r;
    r[0] = (f16)a.x; r[1] = (f16)a.y; r[2] = (f16)a.z; r[3] = (f16)a.w;
    r[4] = (f16)b.x; r[5] = (f16)b.y; r[6] = (f16)b.z; r[7] = (f16)b.w;
    return r;
}

// Stage one 64-key block (K and V) into LDS in (permuted) A-fragment order.
//   K frag fid=kt*2+kc : lane ln half h = K[kt*16+ln][kc*32 + 16*lg + h]
//   V frag fid=dt*2+kk : lane ln half h = V[kk*32 + 8*lg + (h&7) + 16*(h>>3)][dt*16+ln]
__device__ __forceinline__ void stage(const float* __restrict__ kA,  // K row, lane0
                                      const float* __restrict__ kB,  // K row, lane1
                                      const float* __restrict__ vP,  // V col base (dim pair)
                                      v8h* __restrict__ dK, v8h* __restrict__ dV,
                                      int fid, int lane0)
{
    // ---- K: 16 contiguous dims per lane -> 8 f32x4 loads, 4 b128 stores -----
    v8h klo0 = cvt8(*(const f4*)(kA + 0), *(const f4*)(kA + 4));
    v8h khi0 = cvt8(*(const f4*)(kA + 8), *(const f4*)(kA + 12));
    v8h klo1 = cvt8(*(const f4*)(kB + 0), *(const f4*)(kB + 4));
    v8h khi1 = cvt8(*(const f4*)(kB + 8), *(const f4*)(kB + 12));
    dK[fid * 64 +      lane0]     = klo0;
    dK[fid * 64 + 32 + lane0]     = khi0;
    dK[fid * 64 +      lane0 + 1] = klo1;
    dK[fid * 64 + 32 + lane0 + 1] = khi1;

    // ---- V: 16 keys (2 runs of 8) at a dim pair -> 16 f32x2 loads -------------
    f2 e[16];
    #pragma unroll
    for (int i = 0; i < 16; ++i) e[i] = *(const f2*)(vP + (size_t)(i + (i & 8)) * HD);
    v8h a0, a1, b0, b1;
    #pragma unroll
    for (int i = 0; i < 8; ++i) {
        a0[i] = (f16)e[i].x;     b0[i] = (f16)e[i].y;      // halves 0..7
        a1[i] = (f16)e[8 + i].x; b1[i] = (f16)e[8 + i].y;  // halves 8..15
    }
    dV[fid * 64 +      lane0]     = a0;
    dV[fid * 64 + 32 + lane0]     = a1;
    dV[fid * 64 +      lane0 + 1] = b0;
    dV[fid * 64 + 32 + lane0 + 1] = b1;
}

__global__ __launch_bounds__(128)
void fattn_fwd_f16wmma(const float* __restrict__ qw,
                       const float* __restrict__ kw,
                       const float* __restrict__ vw,
                       float* __restrict__ out)
{
    // two v8h halves per lane: 16B/lane stride -> full 64-bank coverage per
    // ds_load_b128 phase (conflict-free).
    __shared__ v8h ldsK[2][8 * 2 * 32];   // 16 KB double-buffered
    __shared__ v8h ldsV[2][8 * 2 * 32];   // 16 KB

    const int tid  = threadIdx.x;
    const int lane = tid & 31;
    const int wave = tid >> 5;
    const int lg   = lane >> 4;
    const int ln   = lane & 15;

    const int ntiles = S_ / QTILE;              // 32 query tiles per (b,h)
    const int bh = blockIdx.x / ntiles;
    const int b  = bh / H_;
    const int h  = bh % H_;
    const int q0 = (blockIdx.x % ntiles) * QTILE + wave * 16;

    const size_t kvBase = (size_t)(b * S_) * HD + h * D_;

    // ---- per-thread staging bases ------------------------------------------
    const int fid   = tid >> 4;                 // 0..7
    const int s     = tid & 15;
    const int lane0 = 2 * s;                    // even lane of pair (lg shared)
    const int lgp   = lane0 >> 4;
    const int ln0   = lane0 & 15;
    const int kt = fid >> 1, kc = fid & 1;      // K frag coords
    const int dt = fid >> 1, kk = fid & 1;      // V frag coords
    const float* kA = kw + kvBase + (size_t)(kt * 16 + ln0)     * HD + kc * 32 + lgp * 16;
    const float* kB = kw + kvBase + (size_t)(kt * 16 + ln0 + 1) * HD + kc * 32 + lgp * 16;
    const float* vP = vw + kvBase + (size_t)(kk * 32 + lgp * 8) * HD + dt * 16 + ln0;

    // ---------------- Q^T B-fragments, pre-scaled by log2(e)/sqrt(D) ----------
    // B half h: dim = c*32 + 16*lg + h
    const float scale = 0.125f * 1.44269504088896f;   // fold exp->exp2 domain
    v16h bQ[2];
    {
        const float* qrow = qw + (size_t)(b * S_ + q0 + ln) * HD + h * D_ + 16 * lg;
        #pragma unroll
        for (int c = 0; c < 2; ++c) {
            f4 a0 = *(const f4*)(qrow + c * 32 + 0);
            f4 a1 = *(const f4*)(qrow + c * 32 + 4);
            f4 a2 = *(const f4*)(qrow + c * 32 + 8);
            f4 a3 = *(const f4*)(qrow + c * 32 + 12);
            bQ[c][0]  = (f16)(a0.x * scale); bQ[c][1]  = (f16)(a0.y * scale);
            bQ[c][2]  = (f16)(a0.z * scale); bQ[c][3]  = (f16)(a0.w * scale);
            bQ[c][4]  = (f16)(a1.x * scale); bQ[c][5]  = (f16)(a1.y * scale);
            bQ[c][6]  = (f16)(a1.z * scale); bQ[c][7]  = (f16)(a1.w * scale);
            bQ[c][8]  = (f16)(a2.x * scale); bQ[c][9]  = (f16)(a2.y * scale);
            bQ[c][10] = (f16)(a2.z * scale); bQ[c][11] = (f16)(a2.w * scale);
            bQ[c][12] = (f16)(a3.x * scale); bQ[c][13] = (f16)(a3.y * scale);
            bQ[c][14] = (f16)(a3.z * scale); bQ[c][15] = (f16)(a3.w * scale);
        }
    }

    // outT accumulators: 4 fragments (dim tiles), element (m=dim, n=query)
    v8f acc[4];
    #pragma unroll
    for (int d = 0; d < 4; ++d)
        #pragma unroll
        for (int r = 0; r < 8; ++r) acc[d][r] = 0.0f;

    float m_i = -1e30f, l_i = 0.0f;   // per-query softmax state (log2 domain)

    // prologue: stage block 0 into buffer 0
    stage(kA, kB, vP, ldsK[0], ldsV[0], fid, lane0);
    __syncthreads();

    kA += (size_t)KBLK * HD; kB += (size_t)KBLK * HD; vP += (size_t)KBLK * HD;

    int cur = 0;
    for (int j0 = 0; j0 < S_; j0 += KBLK, cur ^= 1) {
        // ---- prefetch next block into the other buffer ----------------------
        if (j0 + KBLK < S_) {
            stage(kA, kB, vP, ldsK[cur ^ 1], ldsV[cur ^ 1], fid, lane0);
            kA += (size_t)KBLK * HD; kB += (size_t)KBLK * HD; vP += (size_t)KBLK * HD;
        }

        // ---- S^T = K * Q^T : 4 C-fragments (16 keys x 16 queries), log2 dom --
        v8f st[4];
        #pragma unroll
        for (int t = 0; t < 4; ++t) {
            v8f c;
            #pragma unroll
            for (int r = 0; r < 8; ++r) c[r] = 0.0f;
            #pragma unroll
            for (int d = 0; d < 2; ++d) {
                Frag16 aK;
                aK.h[0] = ldsK[cur][(t * 2 + d) * 64 +      lane];
                aK.h[1] = ldsK[cur][(t * 2 + d) * 64 + 32 + lane];
                c = __builtin_amdgcn_wmma_f32_16x16x32_f16(
                        false, aK.v, false, bQ[d], (short)0, c, false, false);
            }
            st[t] = c;
        }

        // ---- online softmax: keys on (reg, lane-group) axes -----------------
        float mx = st[0][0];
        #pragma unroll
        for (int t = 0; t < 4; ++t)
            #pragma unroll
            for (int r = 0; r < 8; ++r) mx = fmaxf(mx, st[t][r]);
        mx = fmaxf(mx, __shfl_xor(mx, 16, 32));            // combine lane groups

        float mnew = fmaxf(m_i, mx);
        if (__any(mx > m_i)) {          // wave-uniform: rescale only when needed
            float alpha = EXP2(m_i - mnew);
            l_i *= alpha;
            #pragma unroll
            for (int d = 0; d < 4; ++d) acc[d] *= alpha;   // vector op
        }
        m_i = mnew;

        float rs = 0.0f;
        #pragma unroll
        for (int t = 0; t < 4; ++t) {
            v8f dlt = st[t] - mnew;                        // vector sub
            #pragma unroll
            for (int r = 0; r < 8; ++r) {
                float p = EXP2(dlt[r]);
                st[t][r] = p;
                rs += p;
            }
        }
        rs += __shfl_xor(rs, 16, 32);
        l_i += rs;

        // ---- P^T -> B fragments: PURELY LOCAL under pi-permutation ----------
        // bP[c][h] : h<8 -> st[2c][h] ; h>=8 -> st[2c+1][h-8]
        v16h bP[2];
        #pragma unroll
        for (int c = 0; c < 2; ++c)
            #pragma unroll
            for (int r = 0; r < 8; ++r) {
                bP[c][r]     = (f16)st[2 * c][r];
                bP[c][r + 8] = (f16)st[2 * c + 1][r];
            }

        // ---- out^T += V^T * P^T --------------------------------------------
        #pragma unroll
        for (int d = 0; d < 4; ++d)
            #pragma unroll
            for (int c = 0; c < 2; ++c) {
                Frag16 aV;
                aV.h[0] = ldsV[cur][(d * 2 + c) * 64 +      lane];
                aV.h[1] = ldsV[cur][(d * 2 + c) * 64 + 32 + lane];
                acc[d] = __builtin_amdgcn_wmma_f32_16x16x32_f16(
                             false, aV.v, false, bP[c], (short)0, acc[d], false, false);
            }

        __syncthreads();   // staging(next) complete, frags(cur) consumed
    }

    // ---------------- normalize and store [B,S,H,D] ---------------------------
    float inv = 1.0f / l_i;
    float* orow = out + (size_t)((b * S_ + q0 + ln) * H_ + h) * D_;
    #pragma unroll
    for (int d = 0; d < 4; ++d) {
        f4 s0, s1;
        s0.x = acc[d][0] * inv; s0.y = acc[d][1] * inv;
        s0.z = acc[d][2] * inv; s0.w = acc[d][3] * inv;
        s1.x = acc[d][4] * inv; s1.y = acc[d][5] * inv;
        s1.z = acc[d][6] * inv; s1.w = acc[d][7] * inv;
        *(f4*)(orow + d * 16 + (lg << 3))     = s0;
        *(f4*)(orow + d * 16 + (lg << 3) + 4) = s1;
    }
}

extern "C" void kernel_launch(void* const* d_in, const int* in_sizes, int n_in,
                              void* d_out, int out_size, void* d_ws, size_t ws_size,
                              hipStream_t stream)
{
    (void)in_sizes; (void)n_in; (void)out_size; (void)d_ws; (void)ws_size;
    const float* qw = (const float*)d_in[0];
    const float* kw = (const float*)d_in[1];
    const float* vw = (const float*)d_in[2];
    float* out = (float*)d_out;

    dim3 grid(B_ * H_ * (S_ / QTILE));   // 2048 workgroups
    dim3 block(128);                     // 4 waves, 16 query rows each
    fattn_fwd_f16wmma<<<grid, block, 0, stream>>>(qw, kw, vw, out);
}
